// LinearShift_45440753992430
// MI455X (gfx1250) — matrix operand order repaired and checked by reading, not compiled, not run on hardware
//
#include <hip/hip_runtime.h>
#include <hip/hip_bf16.h>

typedef _Float16 v16h __attribute__((ext_vector_type(16)));
typedef _Float16 v8h  __attribute__((ext_vector_type(8)));
typedef _Float16 v4h  __attribute__((ext_vector_type(4)));
typedef float    v8f  __attribute__((ext_vector_type(8)));
typedef float    v4f  __attribute__((ext_vector_type(4)));

// Explicit global-address-space pointers: global_load/global_store codegen.
typedef const v8h  __attribute__((address_space(1)))* gv8h_cp;
typedef const v4f  __attribute__((address_space(1)))* gv4f_cp;
typedef v4h        __attribute__((address_space(1)))* gv4h_p;
typedef const float __attribute__((address_space(1)))* gf32_cp;
typedef float      __attribute__((address_space(1)))* gf32_p;
typedef const char __attribute__((address_space(1)))* gbyte_cp;

#define AS1(T, p) ((T)(unsigned long long)(p))

#define M_DIM 8192
#define K_DIM 4096
#define N_DIM 4096

// round_to_fixed with BITS=16: delta = 2^-8, bound = 2^15
__device__ __forceinline__ float quant_fixed(float v) {
    float r = floorf(fmaf(v, 256.0f, 0.5f));
    r = fminf(fmaxf(r, -32768.0f), 32767.0f);
    return r * 0.00390625f; // * 2^-8
}

// --- Prep: quantize x to fp16 (exact: multiples of 2^-8, |x|<8 => <=11 bits) ---
__global__ __launch_bounds__(256) void quant_x_kernel(const float* __restrict__ x,
                                                      _Float16* __restrict__ xh) {
    unsigned i = blockIdx.x * 256u + threadIdx.x;
    v4f v = AS1(gv4f_cp, x)[i];
    v4h o;
#pragma unroll
    for (int j = 0; j < 4; ++j) o[j] = (_Float16)quant_fixed(v[j]);
    AS1(gv4h_p, xh)[i] = o;
}

// --- Prep: W = (-1)^round(sign) * 2^round(shift), exact power of two in fp16 ---
__global__ __launch_bounds__(256) void make_w_kernel(const float* __restrict__ shift,
                                                     const float* __restrict__ sign,
                                                     _Float16* __restrict__ wh) {
    unsigned i = blockIdx.x * 256u + threadIdx.x;
    v4f sh = AS1(gv4f_cp, shift)[i];
    v4f sg = AS1(gv4f_cp, sign)[i];
    v4h o;
#pragma unroll
    for (int j = 0; j < 4; ++j) {
        int e = (int)rintf(sh[j]);
        float s = 1.0f - 2.0f * rintf(sg[j]);
        o[j] = (_Float16)ldexpf(s, e);
    }
    AS1(gv4h_p, wh)[i] = o;
}

__global__ __launch_bounds__(256) void quant_bias_kernel(const float* __restrict__ bias,
                                                         float* __restrict__ bq) {
    unsigned i = blockIdx.x * 256u + threadIdx.x;
    AS1(gf32_p, bq)[i] = quant_fixed(AS1(gf32_cp, bias)[i]);
}

// --- WMMA GEMM: C[M,N] = Xh[M,K] @ Wh[N,K]^T + bq[N] ---
// Block tile 256x128 (M x N), 256 threads = 8 waves (wave32), wave grid 4(M)x2(N).
// Each wave computes a 64x64 tile = 4x4 WMMA 16x16 tiles:
// 16 WMMAs per K-step against 16 b128 loads (1.0 loads/WMMA).
__global__ __launch_bounds__(256) void gemm_wmma_kernel(const _Float16* __restrict__ Xh,
                                                        const _Float16* __restrict__ Wh,
                                                        const float* __restrict__ bq,
                                                        float* __restrict__ C) {
    const int tid  = threadIdx.x;
    const int wave = tid >> 5;
    const int lane = tid & 31;
    const int row  = lane & 15;   // M-row (A) / N-col (B,C) within 16x16 tile
    const int hi   = lane >> 4;   // which K-half this lane holds

    const int bm = blockIdx.y * 256;
    const int bn = blockIdx.x * 128;
    const int m0 = bm + (wave >> 1) * 64;  // wave M origin (4 tiles of 16)
    const int n0 = bn + (wave & 1) * 64;   // wave N origin (4 tiles of 16)

    gbyte_cp Xb = AS1(gbyte_cp, Xh);
    gbyte_cp Wb = AS1(gbyte_cp, Wh);

    // Byte offsets (avoid 64-bit lshl_add per load; 32-bit dual-issued updates).
    // A fragment (16-bit 16x32, ISA 7.12.2): lane<16 holds M=row, K in {0..7,16..23};
    // lane>=16 holds M=row, K in {8..15,24..31}. Two 16B chunks at +0 and +32 bytes.
    unsigned aOff[4];
#pragma unroll
    for (int mt = 0; mt < 4; ++mt)
        aOff[mt] = ((unsigned)(m0 + mt * 16 + row) * K_DIM + hi * 8) * 2u;

    // B fragment (16-bit 32x16): lane<16 holds col N=row with K in {0..7,16..23}; lane>=16
    // holds K in {8..15,24..31}. B[k,n] = W[n,k]: two 16B chunks at +0 and +32 bytes of row n of W.
    unsigned bOff[4];
#pragma unroll
    for (int nt = 0; nt < 4; ++nt)
        bOff[nt] = ((unsigned)(n0 + nt * 16 + row) * K_DIM + hi * 8) * 2u;

    v8f acc[4][4];
#pragma unroll
    for (int mt = 0; mt < 4; ++mt)
#pragma unroll
        for (int nt = 0; nt < 4; ++nt) acc[mt][nt] = (v8f)(0.0f);

    for (int k = 0; k < K_DIM; k += 32) {
        v16h a[4], b[4];
#pragma unroll
        for (int mt = 0; mt < 4; ++mt) {
            ((v8h*)&a[mt])[0] = *(gv8h_cp)(Xb + aOff[mt]);
            ((v8h*)&a[mt])[1] = *(gv8h_cp)(Xb + aOff[mt] + 32);
            aOff[mt] += 64;
        }
#pragma unroll
        for (int nt = 0; nt < 4; ++nt) {
            ((v8h*)&b[nt])[0] = *(gv8h_cp)(Wb + bOff[nt]);
            ((v8h*)&b[nt])[1] = *(gv8h_cp)(Wb + bOff[nt] + 32);
            bOff[nt] += 64;
        }
#pragma unroll
        for (int mt = 0; mt < 4; ++mt)
#pragma unroll
            for (int nt = 0; nt < 4; ++nt)
                acc[mt][nt] = __builtin_amdgcn_wmma_f32_16x16x32_f16(
                    false, a[mt], false, b[nt], (short)0, acc[mt][nt], false, false);
    }

    // Epilogue. C/D layout: lanes 0-15: M=r, N=lane; lanes 16-31: M=r+8, N=lane-16.
    gf32_p  Cg = AS1(gf32_p, C);
    gf32_cp Bg = AS1(gf32_cp, bq);
#pragma unroll
    for (int nt = 0; nt < 4; ++nt) {
        const unsigned n = (unsigned)(n0 + nt * 16 + row);
        const float bb = Bg[n];
#pragma unroll
        for (int mt = 0; mt < 4; ++mt) {
            const unsigned mbase = (unsigned)(m0 + mt * 16 + hi * 8);
#pragma unroll
            for (int r = 0; r < 8; ++r) {
                Cg[(mbase + r) * (unsigned)N_DIM + n] = acc[mt][nt][r] + bb;
            }
        }
    }
}

extern "C" void kernel_launch(void* const* d_in, const int* in_sizes, int n_in,
                              void* d_out, int out_size, void* d_ws, size_t ws_size,
                              hipStream_t stream) {
    const float* x     = (const float*)d_in[0];
    const float* shift = (const float*)d_in[1];
    const float* sign  = (const float*)d_in[2];
    const float* bias  = (const float*)d_in[3];

    _Float16* Xh = (_Float16*)d_ws;                                      // 64 MB
    _Float16* Wh = (_Float16*)((char*)d_ws + (size_t)M_DIM * K_DIM * 2); // +32 MB
    float*    bq = (float*)((char*)d_ws + (size_t)M_DIM * K_DIM * 2
                                        + (size_t)N_DIM * K_DIM * 2);    // +16 KB

    quant_x_kernel<<<(M_DIM * (size_t)K_DIM / 4) / 256, 256, 0, stream>>>(x, Xh);
    make_w_kernel<<<(N_DIM * (size_t)K_DIM / 4) / 256, 256, 0, stream>>>(shift, sign, Wh);
    quant_bias_kernel<<<N_DIM / 256, 256, 0, stream>>>(bias, bq);

    dim3 grid(N_DIM / 128, M_DIM / 256);
    gemm_wmma_kernel<<<grid, 256, 0, stream>>>(Xh, Wh, bq, (float*)d_out);
}
